// FastConformerMultiHeadAttention_40441412059385
// MI455X (gfx1250) — matrix operand order repaired and checked by reading, not compiled, not run on hardware
//
#include <hip/hip_runtime.h>
#include <hip/hip_bf16.h>
#include <stdint.h>

typedef __bf16 bf16_t;
typedef bf16_t bf16x16 __attribute__((ext_vector_type(16)));
typedef bf16_t bf16x8  __attribute__((ext_vector_type(8)));
typedef float  f32x8   __attribute__((ext_vector_type(8)));

#define WMMA_BF16(a, b, c) \
  __builtin_amdgcn_wmma_f32_16x16x32_bf16(false, (a), false, (b), (short)0, (c), false, false)

static constexpr int BB  = 4;
static constexpr int SS  = 1024;
static constexpr int DD  = 1024;
static constexpr int HH  = 16;
static constexpr int DK  = 64;
static constexpr int P2  = 2048;   // pos rows padded to 2048 (row 2047 == 0)

__device__ __forceinline__ f32x8 zero8() {
  f32x8 z;
#pragma unroll
  for (int i = 0; i < 8; ++i) z[i] = 0.0f;
  return z;
}

// CDNA5 async copy: global -> LDS, 16 bytes, tracked by ASYNCcnt (no VGPR staging).
// VDST = LDS byte address (low 32 bits of generic shared address), VADDR = 64-bit global.
__device__ __forceinline__ void async_b128(const bf16_t* g, bf16_t* l) {
  asm volatile("global_load_async_to_lds_b128 %0, %1, off"
               :: "v"((uint32_t)(uintptr_t)l),
                  "v"((unsigned long long)(uintptr_t)g)
               : "memory");
}

// A fragment (16x32, 16-bit): lane<16 -> M=lane, K={0..7,16..23}; lane>=16 -> K={8..15,24..31}
__device__ __forceinline__ bf16x16 load_a_frag(const bf16_t* base, int ld) {
  const int lane = threadIdx.x & 31;
  const int m = lane & 15, sel = lane >> 4;
  const bf16_t* p = base + (size_t)m * ld + sel * 8;
  bf16x8 lo = *reinterpret_cast<const bf16x8*>(p);
  bf16x8 hi = *reinterpret_cast<const bf16x8*>(p + 16);
  bf16x16 r;
#pragma unroll
  for (int i = 0; i < 8; ++i) { r[i] = lo[i]; r[i + 8] = hi[i]; }
  return r;
}

// B fragment (32x16, 16-bit) from an N-major ("Bt") matrix: lane&15 = N;
// lanes 0-15 hold K={0..7,16..23}, lanes 16-31 hold K={8..15,24..31}.
__device__ __forceinline__ bf16x16 load_b_frag(const bf16_t* base, int ld) {
  const int lane = threadIdx.x & 31;
  const int n = lane & 15, sel = lane >> 4;
  const bf16_t* p = base + (size_t)n * ld + sel * 8;
  bf16x8 lo = *reinterpret_cast<const bf16x8*>(p);
  bf16x8 hi = *reinterpret_cast<const bf16x8*>(p + 16);
  bf16x16 r;
#pragma unroll
  for (int i = 0; i < 8; ++i) { r[i] = lo[i]; r[i + 8] = hi[i]; }
  return r;
}

// ---------------- f32 -> bf16 convert (with zero tail padding) ----------------
__global__ void conv_bf16_kernel(const float* __restrict__ src, bf16_t* __restrict__ dst,
                                 long nsrc, long ntot) {
  long i = (long)blockIdx.x * blockDim.x + threadIdx.x;
  if (i < ntot) dst[i] = (i < nsrc) ? (bf16_t)src[i] : (bf16_t)0.0f;
}

// ---------------- f32 [K,N] -> bf16 [N,K] transpose ----------------
__global__ void transpose_bf16_kernel(const float* __restrict__ src, bf16_t* __restrict__ dst,
                                      int K, int N) {
  __shared__ float tile[32][33];
  const int n0 = blockIdx.x * 32, k0 = blockIdx.y * 32;
  const int tx = threadIdx.x, ty = threadIdx.y;
#pragma unroll
  for (int i = 0; i < 32; i += 8)
    tile[ty + i][tx] = src[(size_t)(k0 + ty + i) * N + n0 + tx];
  __syncthreads();
#pragma unroll
  for (int i = 0; i < 32; i += 8)
    dst[(size_t)(n0 + ty + i) * K + k0 + tx] = (bf16_t)tile[tx][ty + i];
}

// ---------------- WMMA GEMM: out = A[M,Kd] @ Bt[N,Kd]^T + bias, mode-specific store ----
// B tile (shared by all 4 waves) is staged via async-to-LDS, double-buffered.
// mode 0: Q  -> out0 = Q+pos_bias_u, out1 = Q+pos_bias_v, layout [B,H,S,DK] bf16
// mode 1: K  -> out0 layout [B,H,S,DK] bf16
// mode 2: V  -> out0 layout [B,H,DK,S] bf16 (transposed per head)
// mode 3: Pp -> out0 layout [H,2048,DK] bf16 (no bias)
// mode 4: O  -> fout f32 [M, 1024]
__global__ __launch_bounds__(128) void gemm_wmma_kernel(
    const bf16_t* __restrict__ A, const bf16_t* __restrict__ Bt, int Kd,
    const float* __restrict__ bias, const float* __restrict__ ub, const float* __restrict__ vb,
    bf16_t* __restrict__ out0, bf16_t* __restrict__ out1, float* __restrict__ fout, int mode) {
  __shared__ bf16_t Bsh[2][64][32];   // double-buffered 64(N) x 32(K) tile, 2 x 4KB

  const int wave = threadIdx.x >> 5;
  const int m0 = blockIdx.x * 64 + wave * 16;
  const int n0 = blockIdx.y * 64;

  auto stage_b = [&](int kk, int buf) {
#pragma unroll
    for (int j = 0; j < 2; ++j) {
      const int c = (int)threadIdx.x * 2 + j;      // 0..255 16B chunks
      const int r = c >> 2, col = c & 3;           // r: 0..63 (N), col: 0..3 (8 bf16)
      async_b128(Bt + (size_t)(n0 + r) * Kd + kk + col * 8, &Bsh[buf][r][col * 8]);
    }
  };

  f32x8 acc[4];
#pragma unroll
  for (int t = 0; t < 4; ++t) acc[t] = zero8();

  stage_b(0, 0);
  bf16x16 a_cur = load_a_frag(A + (size_t)m0 * Kd, Kd);

  for (int kk = 0; kk < Kd; kk += 32) {
    const int cur = (kk >> 5) & 1, nxt = cur ^ 1;
    const int kk2 = (kk + 32 < Kd) ? kk + 32 : 0;
    __syncthreads();                 // all waves done reading Bsh[nxt] (prev use)
    stage_b(kk2, nxt);               // prefetch next K-step
    bf16x16 a_nxt = load_a_frag(A + (size_t)m0 * Kd + kk2, Kd);
    asm volatile("s_wait_asynccnt 2" ::: "memory");  // older batch (Bsh[cur]) done
    __syncthreads();                 // every wave's Bsh[cur] chunks visible
#pragma unroll
    for (int t = 0; t < 4; ++t) {
      bf16x16 b = load_b_frag(&Bsh[cur][t * 16][0], 32);
      acc[t] = WMMA_BF16(a_cur, b, acc[t]);
    }
    a_cur = a_nxt;
  }

  const int lane = threadIdx.x & 31;
  const int n = lane & 15, sel = lane >> 4;
#pragma unroll
  for (int t = 0; t < 4; ++t) {
    const int ncol = n0 + t * 16 + n;
    const float bs  = bias ? bias[ncol] : 0.0f;
    const float ubv = ub ? ub[ncol] : 0.0f;
    const float vbv = vb ? vb[ncol] : 0.0f;
#pragma unroll
    for (int r = 0; r < 8; ++r) {
      const int mr = m0 + r + 8 * sel;
      const float val = acc[t][r] + bs;
      if (mode == 4) {
        fout[(size_t)mr * DD + ncol] = val;
      } else if (mode == 3) {
        const int hh = ncol >> 6, dk = ncol & 63;
        out0[((size_t)hh * P2 + mr) * DK + dk] = (bf16_t)val;
      } else {
        const int bb = mr >> 10, s = mr & 1023;
        const int hh = ncol >> 6, dk = ncol & 63;
        if (mode == 2) {
          out0[(((size_t)(bb * HH + hh) * DK + dk) * SS + s)] = (bf16_t)val;
        } else {
          const size_t idx = (((size_t)(bb * HH + hh) * SS + s) * DK + dk);
          if (mode == 0) {
            out0[idx] = (bf16_t)(val + ubv);
            out1[idx] = (bf16_t)(val + vbv);
          } else {
            out0[idx] = (bf16_t)val;
          }
        }
      }
    }
  }
}

// ---------------- fused rel-pos flash attention ----------------
// grid (S/64, H, B), block 128 (4 waves); each wave owns 16 query rows.
// K/V tiles (identical across waves) staged via async-to-LDS, double-buffered.
__global__ __launch_bounds__(128) void attn_kernel(
    const bf16_t* __restrict__ Qu, const bf16_t* __restrict__ Qv,
    const bf16_t* __restrict__ Kb, const bf16_t* __restrict__ Vt,
    const bf16_t* __restrict__ Pp, const uint8_t* __restrict__ kmask,
    bf16_t* __restrict__ ctx) {
  struct alignas(16) WaveSmem {
    float  sc[16][32];    // AC score tile
    float  cbd[16][48];   // BD band tile (rel-shift gathered from here)
    float  red[32];       // cross-lane reduction scratch
    float  mrow[16], lrow[16], fac[16];
    bf16_t prob[16][32];  // probs repacked as a WMMA A fragment source
  };
  __shared__ WaveSmem sm[4];
  __shared__ bf16_t Ksh[2][32][64];   // 32 keys x 64 dk, double-buffered (2 x 4KB)
  __shared__ bf16_t Vsh[2][64][32];   // 64 dk x 32 keys, double-buffered (2 x 4KB)

  const int wave = threadIdx.x >> 5;
  const int lane = threadIdx.x & 31;
  WaveSmem& w = sm[wave];
  const int bz = blockIdx.z, h = blockIdx.y;
  const int q0 = blockIdx.x * 64 + wave * 16;
  const int n = lane & 15, sel = lane >> 4;
  const int row = n, half = sel;

  if (lane < 16) { w.mrow[lane] = -3.0e38f; w.lrow[lane] = 0.0f; }
  asm volatile("s_wait_dscnt 0" ::: "memory");

  const bf16_t* qub = Qu + (((size_t)(bz * HH + h) * SS + q0) * DK);
  const bf16_t* qvb = Qv + (((size_t)(bz * HH + h) * SS + q0) * DK);
  const bf16x16 au0 = load_a_frag(qub, DK);
  const bf16x16 au1 = load_a_frag(qub + 32, DK);
  const bf16x16 av0 = load_a_frag(qvb, DK);
  const bf16x16 av1 = load_a_frag(qvb + 32, DK);

  const bf16_t* kbase = Kb + ((size_t)(bz * HH + h) * SS) * DK;
  const bf16_t* vbase = Vt + ((size_t)(bz * HH + h) * DK) * SS;
  const bf16_t* pbase = Pp + (size_t)h * P2 * DK;
  const uint8_t* mb = kmask + (size_t)bz * SS * SS;  // row q=0 (mask is q-broadcast)

  // stage K/V tile for key block kt0 into buffer `buf` (4 async b128 per thread)
  auto stage_kv = [&](int kt0, int buf) {
    const int tid = (int)threadIdx.x;
#pragma unroll
    for (int j = 0; j < 2; ++j) {
      const int c = tid * 2 + j;                 // 0..255
      const int r = c >> 3, col = c & 7;         // K tile: 32 rows x 8 chunks
      async_b128(kbase + (size_t)(kt0 + r) * DK + col * 8, &Ksh[buf][r][col * 8]);
    }
#pragma unroll
    for (int j = 0; j < 2; ++j) {
      const int c = tid * 2 + j;
      const int r = c >> 2, col = c & 3;         // V tile: 64 rows x 4 chunks
      async_b128(vbase + (size_t)r * SS + kt0 + col * 8, &Vsh[buf][r][col * 8]);
    }
  };

  f32x8 o0 = zero8(), o1 = zero8(), o2 = zero8(), o3 = zero8();
  const float scale = 0.125f;  // 1/sqrt(64)

  stage_kv(0, 0);

  for (int k0 = 0; k0 < SS; k0 += 32) {
    const int cur = (k0 >> 5) & 1, nxt = cur ^ 1;
    __syncthreads();                                   // done reading buf[nxt]
    stage_kv((k0 + 32) & (SS - 1), nxt);               // prefetch next key block
    asm volatile("s_wait_asynccnt 4" ::: "memory");    // older batch (buf[cur]) done
    __syncthreads();                                   // all waves' chunks visible

    // ---- AC: (Q+u) @ K^T, two 16-wide tiles, B-fragments from LDS
#pragma unroll
    for (int kt = 0; kt < 2; ++kt) {
      f32x8 acc = zero8();
      bf16x16 b0 = load_b_frag(&Ksh[cur][kt * 16][0], 64);
      acc = WMMA_BF16(au0, b0, acc);
      bf16x16 b1 = load_b_frag(&Ksh[cur][kt * 16][0] + 32, 64);
      acc = WMMA_BF16(au1, b1, acc);
#pragma unroll
      for (int r = 0; r < 8; ++r) w.sc[r + 8 * sel][kt * 16 + n] = acc[r];
    }
    // ---- BD band: (Q+v) @ Pproj^T over p = base + t, t in [0,47]
    const int base = k0 - q0 + SS - 16;  // >= 0, base+47 <= 2047 (padded row = 0)
#pragma unroll
    for (int tt = 0; tt < 3; ++tt) {
      f32x8 acc = zero8();
      bf16x16 b0 = load_b_frag(pbase + (size_t)(base + tt * 16) * DK, DK);
      acc = WMMA_BF16(av0, b0, acc);
      bf16x16 b1 = load_b_frag(pbase + (size_t)(base + tt * 16) * DK + 32, DK);
      acc = WMMA_BF16(av1, b1, acc);
#pragma unroll
      for (int r = 0; r < 8; ++r) w.cbd[r + 8 * sel][tt * 16 + n] = acc[r];
    }
    asm volatile("s_wait_dscnt 0" ::: "memory");

    // ---- online softmax: lane (row, half) owns 16 columns of its row
    float sv[16];
    float lmax = -3.0e38f;
#pragma unroll
    for (int j = 0; j < 16; ++j) {
      const int kj = half * 16 + j;
      float s = (w.sc[row][kj] + w.cbd[row][kj + 15 - row]) * scale;  // rel-shift gather
      if (mb[k0 + kj]) s = -1.0e9f;
      sv[j] = s;
      lmax = fmaxf(lmax, s);
    }
    w.red[lane] = lmax;
    asm volatile("s_wait_dscnt 0" ::: "memory");
    const float rmax = fmaxf(w.red[row], w.red[row + 16]);
    const float mold = w.mrow[row];
    const float mnew = fmaxf(mold, rmax);
    const float factor = __expf(mold - mnew);
    float lsum = 0.0f;
#pragma unroll
    for (int j = 0; j < 16; ++j) {
      const float p = __expf(sv[j] - mnew);
      w.prob[row][half * 16 + j] = (bf16_t)p;
      lsum += p;
    }
    w.red[lane] = lsum;
    asm volatile("s_wait_dscnt 0" ::: "memory");
    if (half == 0) {
      w.lrow[row] = w.lrow[row] * factor + w.red[row] + w.red[row + 16];
      w.mrow[row] = mnew;
      w.fac[row] = factor;
    }
    asm volatile("s_wait_dscnt 0" ::: "memory");

    // ---- rescale accumulators, then O += P @ V (V-fragments from LDS)
#pragma unroll
    for (int r = 0; r < 8; ++r) {
      const float f = w.fac[r + 8 * sel];
      o0[r] *= f; o1[r] *= f; o2[r] *= f; o3[r] *= f;
    }
    const bf16x16 pa = load_a_frag(&w.prob[0][0], 32);
    bf16x16 bv0 = load_b_frag(&Vsh[cur][0][0], 32);
    o0 = WMMA_BF16(pa, bv0, o0);
    bf16x16 bv1 = load_b_frag(&Vsh[cur][16][0], 32);
    o1 = WMMA_BF16(pa, bv1, o1);
    bf16x16 bv2 = load_b_frag(&Vsh[cur][32][0], 32);
    o2 = WMMA_BF16(pa, bv2, o2);
    bf16x16 bv3 = load_b_frag(&Vsh[cur][48][0], 32);
    o3 = WMMA_BF16(pa, bv3, o3);
  }

  asm volatile("s_wait_dscnt 0" ::: "memory");
#pragma unroll
  for (int r = 0; r < 8; ++r) {
    const int qr = q0 + r + 8 * sel;
    const float inv = 1.0f / w.lrow[r + 8 * sel];
    bf16_t* cp = ctx + (size_t)(bz * SS + qr) * DD + h * DK;
    cp[0 * 16 + n] = (bf16_t)(o0[r] * inv);
    cp[1 * 16 + n] = (bf16_t)(o1[r] * inv);
    cp[2 * 16 + n] = (bf16_t)(o2[r] * inv);
    cp[3 * 16 + n] = (bf16_t)(o3[r] * inv);
  }
}

extern "C" void kernel_launch(void* const* d_in, const int* in_sizes, int n_in,
                              void* d_out, int out_size, void* d_ws, size_t ws_size,
                              hipStream_t stream) {
  (void)in_sizes; (void)n_in; (void)out_size; (void)ws_size;
  const float*   X    = (const float*)d_in[0];
  const float*   pos  = (const float*)d_in[1];
  const uint8_t* msk  = (const uint8_t*)d_in[2];
  const float*   Wq   = (const float*)d_in[3];
  const float*   bq   = (const float*)d_in[4];
  const float*   Wk   = (const float*)d_in[5];
  const float*   bk   = (const float*)d_in[6];
  const float*   Wv   = (const float*)d_in[7];
  const float*   bv   = (const float*)d_in[8];
  const float*   Wpos = (const float*)d_in[9];
  const float*   pu   = (const float*)d_in[10];
  const float*   pv   = (const float*)d_in[11];
  const float*   Wo   = (const float*)d_in[12];
  const float*   bo   = (const float*)d_in[13];
  float* out = (float*)d_out;

  char* ws = (char*)d_ws;
  size_t off = 0;
  auto take = [&](size_t bytes) -> char* {
    char* p = ws + off;
    off += (bytes + 255) & ~(size_t)255;
    return p;
  };
  bf16_t* Xb    = (bf16_t*)take((size_t)4096 * 1024 * 2);
  bf16_t* Pb    = (bf16_t*)take((size_t)2048 * 1024 * 2);
  bf16_t* WqT   = (bf16_t*)take((size_t)1024 * 1024 * 2);
  bf16_t* WkT   = (bf16_t*)take((size_t)1024 * 1024 * 2);
  bf16_t* WvT   = (bf16_t*)take((size_t)1024 * 1024 * 2);
  bf16_t* WposT = (bf16_t*)take((size_t)1024 * 1024 * 2);
  bf16_t* WoT   = (bf16_t*)take((size_t)1024 * 1024 * 2);
  bf16_t* Qub   = (bf16_t*)take((size_t)BB * HH * SS * DK * 2);
  bf16_t* Qvb   = (bf16_t*)take((size_t)BB * HH * SS * DK * 2);
  bf16_t* Kbf   = (bf16_t*)take((size_t)BB * HH * SS * DK * 2);
  bf16_t* Vtb   = (bf16_t*)take((size_t)BB * HH * DK * SS * 2);
  bf16_t* Ppj   = (bf16_t*)take((size_t)HH * P2 * DK * 2);
  bf16_t* Ctx   = (bf16_t*)take((size_t)4096 * 1024 * 2);

  // 1) precision conversion + padding
  {
    long nx = 4096L * 1024;
    conv_bf16_kernel<<<(unsigned)((nx + 255) / 256), 256, 0, stream>>>(X, Xb, nx, nx);
    long ptot = 2048L * 1024, psrc = 2047L * 1024;
    conv_bf16_kernel<<<(unsigned)((ptot + 255) / 256), 256, 0, stream>>>(pos, Pb, psrc, ptot);
  }
  // 2) weight transposes (N-major bf16 for WMMA B fragments)
  {
    dim3 tb(32, 8), tg(32, 32);
    transpose_bf16_kernel<<<tg, tb, 0, stream>>>(Wq, WqT, 1024, 1024);
    transpose_bf16_kernel<<<tg, tb, 0, stream>>>(Wk, WkT, 1024, 1024);
    transpose_bf16_kernel<<<tg, tb, 0, stream>>>(Wv, WvT, 1024, 1024);
    transpose_bf16_kernel<<<tg, tb, 0, stream>>>(Wpos, WposT, 1024, 1024);
    transpose_bf16_kernel<<<tg, tb, 0, stream>>>(Wo, WoT, 1024, 1024);
  }
  // 3) projections
  {
    dim3 gb(64, 16);  // M=4096
    gemm_wmma_kernel<<<gb, 128, 0, stream>>>(Xb, WqT, 1024, bq, pu, pv, Qub, Qvb, nullptr, 0);
    gemm_wmma_kernel<<<gb, 128, 0, stream>>>(Xb, WkT, 1024, bk, nullptr, nullptr, Kbf, nullptr, nullptr, 1);
    gemm_wmma_kernel<<<gb, 128, 0, stream>>>(Xb, WvT, 1024, bv, nullptr, nullptr, Vtb, nullptr, nullptr, 2);
    dim3 gp(32, 16);  // M=2048 (padded pos rows)
    gemm_wmma_kernel<<<gp, 128, 0, stream>>>(Pb, WposT, 1024, nullptr, nullptr, nullptr, Ppj, nullptr, nullptr, 3);
  }
  // 4) fused rel-pos flash attention
  {
    dim3 ga(SS / 64, HH, BB);
    attn_kernel<<<ga, 128, 0, stream>>>(Qub, Qvb, Kbf, Vtb, Ppj, msk, Ctx);
  }
  // 5) output projection -> f32 d_out
  {
    dim3 gb(64, 16);
    gemm_wmma_kernel<<<gb, 128, 0, stream>>>(Ctx, WoT, 1024, bo, nullptr, nullptr, nullptr, nullptr, out, 4);
  }
}
